// NativeDenseGATLayer_39737037423327
// MI455X (gfx1250) — matrix operand order repaired and checked by reading, not compiled, not run on hardware
//
#include <hip/hip_runtime.h>

typedef __attribute__((ext_vector_type(16))) _Float16 v16h;
typedef __attribute__((ext_vector_type(8)))  float    v8f;

#define FIN 128
#define FOUT 64
#define NN 2048
#define ALPHA 0.2f

// ---------------------------------------------------------------------------
// Kernel 1: Wh = h @ W (f32 math), f_src = Wh.a[:64], f_dst = Wh.a[64:],
// and store Wh as f16 pre-swizzled into the WMMA B-matrix (32x16, 16-bit)
// register layout: element (K=kin, N=ncol) of chunk (b,chunk,tile) lives at
//   addr = ((b*64+chunk)*4 + tile)*512 + (ncol + 16*((kin/8)%2))*16 + (kin%8) + 8*(kin/16)
// ---------------------------------------------------------------------------
#define KSTEP(HV, KBASE)                                                      \
  _Pragma("unroll")                                                           \
  for (int k2 = 0; k2 < 32; ++k2) {                                           \
    float hk = __shfl(HV, k2, 32);                                            \
    acc0 = fmaf(hk, Wl[(KBASE + k2) * FOUT + lane], acc0);                    \
    acc1 = fmaf(hk, Wl[(KBASE + k2) * FOUT + lane + 32], acc1);               \
  }

__global__ __launch_bounds__(256) void gat_wh_kernel(
    const float* __restrict__ h, const float* __restrict__ W,
    const float* __restrict__ a, _Float16* __restrict__ whB,
    float* __restrict__ f_src, float* __restrict__ f_dst) {
  __shared__ float Wl[FIN * FOUT];
  const int tid = threadIdx.x;
  for (int i = tid * 4; i < FIN * FOUT; i += 256 * 4)
    *(float4*)(Wl + i) = *(const float4*)(W + i);
  __syncthreads();

  const int lane = tid & 31, wave = tid >> 5;
  const float a0 = a[lane], a1 = a[lane + 32];
  const float a2 = a[64 + lane], a3 = a[96 + lane];
  const int gw = blockIdx.x * 8 + wave;  // 4096 waves, 4 rows each

  for (int r = 0; r < 4; ++r) {
    const int row = gw * 4 + r;  // [0, 16384)
    const float* hr = h + (size_t)row * FIN;
    const float hv0 = hr[lane], hv1 = hr[lane + 32];
    const float hv2 = hr[lane + 64], hv3 = hr[lane + 96];
    float acc0 = 0.f, acc1 = 0.f;  // Wh[row][lane], Wh[row][lane+32]
    KSTEP(hv0, 0)
    KSTEP(hv1, 32)
    KSTEP(hv2, 64)
    KSTEP(hv3, 96)

    float ps = acc0 * a0 + acc1 * a1;
    float pd = acc0 * a2 + acc1 * a3;
#pragma unroll
    for (int s = 16; s > 0; s >>= 1) {
      ps += __shfl_xor(ps, s, 32);
      pd += __shfl_xor(pd, s, 32);
    }
    if (lane == 0) { f_src[row] = ps; f_dst[row] = pd; }

    // swizzled f16 stores (B-matrix layout)
    const int m = row & (NN - 1);
    const int chunk = m >> 5, kin = m & 31;
    const int slot = (kin & 7) | ((kin >> 4) << 3), khalf = (kin >> 3) & 1;
    const int b = row >> 11;
    const size_t cbase = ((size_t)((b * 64 + chunk) * 4)) << 9;
    {
      int j = lane, tile = j >> 4, ncol = j & 15;
      whB[cbase + ((size_t)tile << 9) + (size_t)((ncol + (khalf << 4)) << 4) + slot] =
          (_Float16)acc0;
      j = lane + 32; tile = j >> 4; ncol = j & 15;
      whB[cbase + ((size_t)tile << 9) + (size_t)((ncol + (khalf << 4)) << 4) + slot] =
          (_Float16)acc1;
    }
  }
}

// ---------------------------------------------------------------------------
// Kernel 2: fused masked-softmax attention + aggregation.
// Grid: 8 batches * 128 row-tiles = 1024 blocks of 64 threads (2 waves).
// Wave w handles m in [w*1024,(w+1)*1024); per 32-m chunk it builds the
// 16x32 f16 P tile in the WMMA A layout and does 4x v_wmma_f32_16x16x32_f16.
// ---------------------------------------------------------------------------
__global__ __launch_bounds__(64) void gat_attn_kernel(
    const int* __restrict__ adj, const _Float16* __restrict__ whB,
    const float* __restrict__ f_src, const float* __restrict__ f_dst,
    float* __restrict__ out) {
  __shared__ float fdl[NN];
  __shared__ float accred[4 * 8 * 32];
  __shared__ float sumred[32];
  __shared__ float wmaxs[2];

  const int b = blockIdx.x >> 7;
  const int row_base = (blockIdx.x & 127) << 4;
  const int tid = threadIdx.x, lane = tid & 31, wave = tid >> 5;

  // stage f_dst[b,:] into LDS, track max
  const float* fd = f_dst + b * NN;
  float lm = -3.0e38f;
  for (int i = tid; i < NN; i += 64) {
    float v = fd[i];
    fdl[i] = v;
    lm = fmaxf(lm, v);
  }
#pragma unroll
  for (int s = 16; s > 0; s >>= 1) lm = fmaxf(lm, __shfl_xor(lm, s, 32));
  if (lane == 0) wmaxs[wave] = lm;
  __syncthreads();
  const float g = fmaxf(wmaxs[0], wmaxs[1]);

  const int myrow = lane & 15;
  const float fsr = f_src[b * NN + row_base + myrow];
  const float tub = fsr + g;
  const float UB = fmaxf(tub, ALPHA * tub);  // exact row upper bound of e
  const int o = (lane < 16) ? 0 : 8;         // A-layout K offset per lane half
  const int* adjr = adj + (((size_t)b * NN + row_base + myrow) << 11);

  v8f cacc[4] = {};
  float suml = 0.f;

  const int c_begin = wave * 32, c_end = c_begin + 32;
  for (int ch = c_begin; ch < c_end; ++ch) {
    const int m0 = ch << 5;
    const int4* ap = (const int4*)(adjr + m0 + o);
    const int4 A0 = ap[0], A1 = ap[1], A2 = ap[4], A3 = ap[5];
    const float4* fp = (const float4*)(fdl + m0 + o);
    const float4 F0 = fp[0], F1 = fp[1], F2 = fp[4], F3 = fp[5];

    const int av[16] = {A0.x, A0.y, A0.z, A0.w, A1.x, A1.y, A1.z, A1.w,
                        A2.x, A2.y, A2.z, A2.w, A3.x, A3.y, A3.z, A3.w};
    const float fv[16] = {F0.x, F0.y, F0.z, F0.w, F1.x, F1.y, F1.z, F1.w,
                          F2.x, F2.y, F2.z, F2.w, F3.x, F3.y, F3.z, F3.w};
    v16h A;
    float s = 0.f;
#pragma unroll
    for (int i = 0; i < 16; ++i) {
      float t = fsr + fv[i];
      float e = fmaxf(t, ALPHA * t) - UB;      // leaky relu, shifted; e <= 0
      float p = (av[i] > 0) ? __expf(e) : 0.f; // mask
      s += p;
      A[i] = (_Float16)p;
    }
    suml += s;

    const _Float16* wb = whB + (((size_t)((b * 64 + ch) * 4)) << 9) + (lane << 4);
#pragma unroll
    for (int t2 = 0; t2 < 4; ++t2) {
      v16h B = *(const v16h*)(wb + ((size_t)t2 << 9));
      cacc[t2] = __builtin_amdgcn_wmma_f32_16x16x32_f16(
          false, A, false, B, (short)0, cacc[t2], false, false);
    }
  }

  // combine lane halves (disjoint K per half-wave), then waves via LDS
  suml += __shfl_xor(suml, 16, 32);
  if (wave == 1) {
#pragma unroll
    for (int t2 = 0; t2 < 4; ++t2)
#pragma unroll
      for (int r = 0; r < 8; ++r) accred[(t2 * 8 + r) * 32 + lane] = cacc[t2][r];
    sumred[lane] = suml;
  }
  __syncthreads();
  if (wave == 0) {
    const float inv = 1.0f / (suml + sumred[lane]);
#pragma unroll
    for (int r = 0; r < 8; ++r) {
      // C layout: VGPR r -> row r (lanes 0-15), row r+8 (lanes 16-31)
      const float ilo = __shfl(inv, r, 32);
      const float ihi = __shfl(inv, r + 8, 32);
      const float ir = (lane < 16) ? ilo : ihi;
      const int rowg = row_base + r + ((lane < 16) ? 0 : 8);
      float* orow = out + (((size_t)b * NN + rowg) << 6) + (lane & 15);
#pragma unroll
      for (int t2 = 0; t2 < 4; ++t2) {
        float v = (cacc[t2][r] + accred[(t2 * 8 + r) * 32 + lane]) * ir;
        v = (v > 0.f) ? v : (__expf(v) - 1.f);  // ELU
        orow[t2 * 16] = v;
      }
    }
  }
}

extern "C" void kernel_launch(void* const* d_in, const int* in_sizes, int n_in,
                              void* d_out, int out_size, void* d_ws, size_t ws_size,
                              hipStream_t stream) {
  (void)in_sizes; (void)n_in; (void)out_size; (void)ws_size;
  const float* h   = (const float*)d_in[0];  // [8,2048,128]
  const int*   adj = (const int*)d_in[1];    // [8,2048,2048]
  const float* W   = (const float*)d_in[2];  // [128,64]
  const float* a   = (const float*)d_in[3];  // [128,1]
  float* out = (float*)d_out;                // [8,2048,64]

  _Float16* whB  = (_Float16*)d_ws;                      // 2 MB swizzled Wh f16
  float*    fsrc = (float*)((char*)d_ws + (2u << 20));   // 64 KB
  float*    fdst = fsrc + 8 * NN;                        // 64 KB

  gat_wh_kernel<<<512, 256, 0, stream>>>(h, W, a, whB, fsrc, fdst);
  gat_attn_kernel<<<1024, 64, 0, stream>>>(adj, whB, fsrc, fdst, out);
}
